// GNBlock_42133629173928
// MI455X (gfx1250) — compile-verified, matrix operand order checked by reading, not run on hardware
//
#include <hip/hip_runtime.h>
#include <stdint.h>

// ---------------------------------------------------------------------------
// Types / helpers
// ---------------------------------------------------------------------------
typedef __attribute__((ext_vector_type(16))) __bf16 v16bf;
typedef __attribute__((ext_vector_type(8)))  float  v8f;

union FragB16 { v16bf v; uint4 q[2]; unsigned short s[16]; };

__device__ __forceinline__ unsigned short f2bf(float f) {
  unsigned int u = __builtin_bit_cast(unsigned int, f);
  u += 0x7FFFu + ((u >> 16) & 1u);           // round-to-nearest-even
  return (unsigned short)(u >> 16);
}
__device__ __forceinline__ float bf2f(unsigned short h) {
  unsigned int u = ((unsigned int)h) << 16;
  return __builtin_bit_cast(float, u);
}

// ---------------------------------------------------------------------------
// Weight pre-pack: f32 W[N][K] -> bf16 tiles in exact per-lane WMMA-B order.
// Tile (n_tile,k_tile): 32 lanes x 16 bf16 contiguous. Within a lane the 16
// elements map to VGPR pairs: lanes 0-15 hold k = {0..7, 16..23}, lanes 16-31
// hold k = {8..15, 24..31} (mirrors the 16-bit 16x32 A layout).
// ---------------------------------------------------------------------------
__global__ void pack_weights(const float* __restrict__ W,
                             unsigned short* __restrict__ out, int N, int Kd) {
  int idx = blockIdx.x * 256 + threadIdx.x;
  int total = N * Kd;
  if (idx >= total) return;
  int tile = idx >> 9;            // 512 elems / tile
  int r    = idx & 511;
  int lane = r >> 4;
  int j    = r & 15;
  int kt   = Kd >> 5;             // k-tiles
  int n_tile = tile / kt;
  int k_tile = tile % kt;
  int n = n_tile * 16 + (lane & 15);
  int k;
  if (lane < 16) k = (j < 8) ? j       : 16 + (j - 8);
  else           k = (j < 8) ? 8 + j   : 24 + (j - 8);
  k += k_tile * 32;
  out[idx] = f2bf(W[(size_t)n * Kd + k]);
}

// ---------------------------------------------------------------------------
// Fused WMMA GEMM  H = X @ W^T + b  (64 rows x NOUT cols per workgroup)
// LOADER: 0 = edge gather-concat [g_e|src|dst|edge]  (K=512)
//         1 = 3-way concat [idx-gathered | row | row] (K=384)
//         2 = bf16 H_prev with batchnorm scale/shift + ReLU applied (K=256)
// STATS:  accumulate per-column sum / sum-of-squares and store bf16 H.
//         else: write f32 (final layer).
// ---------------------------------------------------------------------------
template <int K, int NOUT, int LOADER, bool STATS>
__global__ void __launch_bounds__(256) gemm_mlp(
    const unsigned short* __restrict__ Wp, const float* __restrict__ bias,
    int B,
    const unsigned short* __restrict__ Xprev,          // LOADER==2
    const float* __restrict__ scale, const float* __restrict__ shift,
    const float* __restrict__ src0, const float* __restrict__ src1,
    const float* __restrict__ src2, const float* __restrict__ src3,
    const int* __restrict__ conn, const int* __restrict__ gidx,
    unsigned short* __restrict__ Hout, float* __restrict__ Fout,
    float* __restrict__ statSum, float* __restrict__ statSq) {

  constexpr int KP = K + 8;                 // +16B pad: conflict-free b128 A loads
  __shared__ unsigned short sX[64 * KP];
  __shared__ float sStat[STATS ? 2 * NOUT : 1];
  __shared__ float sNorm[(LOADER == 2) ? 2 * K : 1];
  __shared__ int   sIdx[(LOADER == 0) ? 192 : ((LOADER == 1) ? 64 : 1)];

  const int tid = threadIdx.x;
  const int blockRow = blockIdx.x * 64;

  if (STATS) {
    for (int j = tid; j < 2 * NOUT; j += 256) sStat[j] = 0.f;
  }

  // ---------------- stage X tile into LDS as bf16 ----------------
  if (LOADER == 2) {
    for (int j = tid; j < K; j += 256) { sNorm[j] = scale[j]; sNorm[K + j] = shift[j]; }
    __syncthreads();
    constexpr int ITERS = (64 * K) / (256 * 8);
    for (int it = 0; it < ITERS; ++it) {
      int id  = it * 256 + tid;
      int r   = id / (K / 8);
      int col = (id % (K / 8)) * 8;
      int row = blockRow + r;
      union { uint4 q; unsigned short s[8]; } o;
      if (row < B) {
        uint4 raw = *(const uint4*)(Xprev + (size_t)row * K + col);
        const unsigned short* rs = (const unsigned short*)&raw;
#pragma unroll
        for (int j = 0; j < 8; ++j) {
          float y = bf2f(rs[j]) * sNorm[col + j] + sNorm[K + col + j];
          o.s[j] = f2bf(y > 0.f ? y : 0.f);
        }
      } else {
        o.q = uint4{0u, 0u, 0u, 0u};
      }
      *(uint4*)&sX[r * KP + col] = o.q;
    }
  } else if (LOADER == 0) {
    if (tid < 64) {
      int e = blockRow + tid;
      int g = 0, s = 0, d = 0;
      if (e < B) { g = gidx[e]; s = conn[2 * e]; d = conn[2 * e + 1]; }
      sIdx[tid] = g; sIdx[64 + tid] = s; sIdx[128 + tid] = d;
    }
    __syncthreads();
#pragma unroll 4
    for (int it = 0; it < 32; ++it) {           // 64 rows x 512 cols, float4 each
      int id  = it * 256 + tid;
      int r   = id >> 7;
      int col = (id & 127) << 2;
      int row = blockRow + r;
      union { uint2 q; unsigned short s[4]; } o;
      if (row < B) {
        const float* p;
        if (col < 128)      p = src0 + (size_t)sIdx[r] * 128 + col;
        else if (col < 256) p = src1 + (size_t)sIdx[64 + r] * 128 + (col - 128);
        else if (col < 384) p = src1 + (size_t)sIdx[128 + r] * 128 + (col - 256);
        else                p = src3 + (size_t)row * 128 + (col - 384);
        float4 f = *(const float4*)p;
        o.s[0] = f2bf(f.x); o.s[1] = f2bf(f.y); o.s[2] = f2bf(f.z); o.s[3] = f2bf(f.w);
      } else {
        o.q = uint2{0u, 0u};
      }
      *(uint2*)&sX[r * KP + col] = o.q;
    }
  } else if (LOADER == 1) {
    if (tid < 64) {
      int n = blockRow + tid;
      int g = 0;
      if (n < B) g = gidx ? gidx[n] : n;
      sIdx[tid] = g;
    }
    __syncthreads();
#pragma unroll 4
    for (int it = 0; it < 24; ++it) {           // 64 rows x 384 cols, float4 each
      int id  = it * 256 + tid;
      int r   = id / 96;
      int col = (id % 96) * 4;
      int row = blockRow + r;
      union { uint2 q; unsigned short s[4]; } o;
      if (row < B) {
        const float* p;
        if (col < 128)      p = src0 + (size_t)sIdx[r] * 128 + col;
        else if (col < 256) p = src1 + (size_t)row * 128 + (col - 128);
        else                p = src2 + (size_t)row * 128 + (col - 256);
        float4 f = *(const float4*)p;
        o.s[0] = f2bf(f.x); o.s[1] = f2bf(f.y); o.s[2] = f2bf(f.z); o.s[3] = f2bf(f.w);
      } else {
        o.q = uint2{0u, 0u};
      }
      *(uint2*)&sX[r * KP + col] = o.q;
    }
  }
  __syncthreads();

  // ---------------- WMMA main loop ----------------
  const int lane  = tid & 31;
  const int wave  = tid >> 5;
  const int mt    = wave >> 1;      // 0..3 : which 16-row sub-tile
  const int nhalf = wave & 1;       // 0..1 : which half of N
  constexpr int NT = NOUT / 32;     // n-tiles per wave (8 for N=256, 4 for N=128)
  constexpr int KT = K / 32;

  v8f acc[NT];
#pragma unroll
  for (int nt = 0; nt < NT; ++nt)
#pragma unroll
    for (int v = 0; v < 8; ++v) acc[nt][v] = 0.f;

  const int arow  = mt * 16 + (lane & 15);
  const int akoff = (lane >> 4) * 8;            // lanes>=16 hold K 8..15 / 24..31

  for (int kt = 0; kt < KT; ++kt) {
    FragB16 a;
    const unsigned short* ap = &sX[arow * KP + kt * 32 + akoff];
    a.q[0] = *(const uint4*)ap;                 // k .. k+7   (VGPR0..3)
    a.q[1] = *(const uint4*)(ap + 16);          // k+16..k+23 (VGPR4..7)
#pragma unroll
    for (int nt = 0; nt < NT; ++nt) {
      int ntg = nhalf * NT + nt;
      const unsigned short* bp = Wp + (((size_t)ntg * KT + kt) * 32 + lane) * 16;
      FragB16 b;
      b.q[0] = *(const uint4*)bp;
      b.q[1] = *(const uint4*)(bp + 8);
      acc[nt] = __builtin_amdgcn_wmma_f32_16x16x32_bf16(
          false, a.v, false, b.v, (short)0, acc[nt], false, false);
    }
  }

  // ---------------- epilogue: bias, stats, store ----------------
  const int rowBase = blockRow + mt * 16 + ((lane >> 4) << 3);
#pragma unroll
  for (int nt = 0; nt < NT; ++nt) {
    const int col = (nhalf * NT + nt) * 16 + (lane & 15);
    const float bcol = bias[col];
    if (STATS) {
      float s = 0.f, q = 0.f;
#pragma unroll
      for (int v = 0; v < 8; ++v) {
        int row = rowBase + v;
        if (row < B) {
          float h = acc[nt][v] + bcol;
          s += h; q += h * h;
          Hout[(size_t)row * NOUT + col] = f2bf(h);
        }
      }
      atomicAdd(&sStat[col], s);               // ds_add_f32
      atomicAdd(&sStat[NOUT + col], q);
    } else {
#pragma unroll
      for (int v = 0; v < 8; ++v) {
        int row = rowBase + v;
        if (row < B) Fout[(size_t)row * NOUT + col] = acc[nt][v] + bcol;
      }
    }
  }
  if (STATS) {
    __syncthreads();
    if (tid < NOUT) {
      atomicAdd(&statSum[tid], sStat[tid]);
      atomicAdd(&statSq[tid],  sStat[NOUT + tid]);
    }
  }
}

// ---------------------------------------------------------------------------
// Small kernels
// ---------------------------------------------------------------------------
__global__ void norm_params(const float* __restrict__ sum, const float* __restrict__ sq,
                            const float* __restrict__ g, const float* __restrict__ be,
                            float* __restrict__ scale, float* __restrict__ shift,
                            float invB, int n) {
  int t = blockIdx.x * blockDim.x + threadIdx.x;
  if (t < n) {
    float m  = sum[t] * invB;
    float v  = sq[t] * invB - m * m;
    float sc = g[t] * rsqrtf(v + 1e-5f);
    scale[t] = sc;
    shift[t] = be[t] - m * sc;
  }
}

__global__ void zero_f32(float* __restrict__ p, long long n) {
  long long i = (long long)blockIdx.x * 256 + threadIdx.x;
  if (i < n) p[i] = 0.f;
}

// per-node segment sum: agg[conn[e,1]] += new_edge[e]
__global__ void scatter_add_rows(const float* __restrict__ src, const int* __restrict__ conn,
                                 float* __restrict__ dst, long long total) {
  long long i = (long long)blockIdx.x * 256 + threadIdx.x;
  if (i >= total) return;
  int e = (int)(i >> 7);
  int c = (int)(i & 127);
  int d = conn[2 * e + 1];
  atomicAdd(&dst[(size_t)d * 128 + c], src[i]);
}

// per-graph segment sum with LDS-private 64x128 accumulator
__global__ void __launch_bounds__(256) scatter_add_graph(
    const float* __restrict__ src, const int* __restrict__ idx,
    float* __restrict__ dst, long long total) {
  __shared__ float lacc[64 * 128];
  for (int j = threadIdx.x; j < 8192; j += 256) lacc[j] = 0.f;
  __syncthreads();
  long long stride = (long long)gridDim.x * 256;
  for (long long i = (long long)blockIdx.x * 256 + threadIdx.x; i < total; i += stride) {
    int e = (int)(i >> 7);
    int c = (int)(i & 127);
    atomicAdd(&lacc[idx[e] * 128 + c], src[i]);
  }
  __syncthreads();
  for (int j = threadIdx.x; j < 8192; j += 256) atomicAdd(&dst[j], lacc[j]);
}

// ---------------------------------------------------------------------------
// Host launcher
// ---------------------------------------------------------------------------
extern "C" void kernel_launch(void* const* d_in, const int* in_sizes, int n_in,
                              void* d_out, int out_size, void* d_ws, size_t ws_size,
                              hipStream_t stream) {
  (void)n_in; (void)out_size; (void)ws_size;
  const float* globalF = (const float*)d_in[0];
  const float* nodeF   = (const float*)d_in[1];
  const float* edgeF   = (const float*)d_in[2];
  const int*   conn    = (const int*)d_in[3];
  const int*   gin     = (const int*)d_in[4];
  const int*   gie     = (const int*)d_in[5];
  const int G  = in_sizes[0] / 128;
  const int Nn = in_sizes[1] / 128;
  const int E  = in_sizes[2] / 128;

  // params pytree (dict keys sorted): blocks edge(0), glob(1), node(2);
  // within block: W0..W3, b0..b3, be0..be2, g0..g2
  const float *W[3][4], *bb[3][4], *beP[3][3], *gaP[3][3];
  for (int blk = 0; blk < 3; ++blk) {
    int base = 6 + 14 * blk;
    for (int i = 0; i < 4; ++i) W[blk][i]   = (const float*)d_in[base + i];
    for (int i = 0; i < 4; ++i) bb[blk][i]  = (const float*)d_in[base + 4 + i];
    for (int i = 0; i < 3; ++i) beP[blk][i] = (const float*)d_in[base + 8 + i];
    for (int i = 0; i < 3; ++i) gaP[blk][i] = (const float*)d_in[base + 11 + i];
  }
  const int BE = 0, BG = 1, BN = 2;           // edge, glob, node

  // outputs (tuple order: new_global, new_node, new_edge)
  float* outGlob = (float*)d_out;
  float* outNode = outGlob + (size_t)G * 128;
  float* outEdge = outNode + (size_t)Nn * 128;

  // workspace carve-up
  char* wsb = (char*)d_ws;
  size_t off = 0;
  auto carve = [&](size_t bytes) -> void* {
    void* p = wsb + off;
    off = (off + bytes + 255) & ~(size_t)255;
    return p;
  };
  // packed weights: layer dims per block
  const int Kdim[3][4] = {{512, 256, 256, 256}, {384, 256, 256, 256}, {384, 256, 256, 256}};
  const int Ndim[4]    = {256, 256, 256, 128};
  unsigned short* pw[3][4];
  for (int blk = 0; blk < 3; ++blk)
    for (int l = 0; l < 4; ++l)
      pw[blk][l] = (unsigned short*)carve((size_t)Ndim[l] * Kdim[blk][l] * 2);

  unsigned short* HA = (unsigned short*)carve((size_t)E * 256 * 2);
  unsigned short* HB = (unsigned short*)carve((size_t)E * 256 * 2);
  float* aggNode = (float*)carve((size_t)Nn * 128 * 4);
  float* aggEG   = (float*)carve((size_t)G * 128 * 4);
  float* aggNG   = (float*)carve((size_t)G * 128 * 4);
  float* stats   = (float*)carve(1024 * 4);
  float* stSum = stats, *stSq = stats + 256, *stScale = stats + 512, *stShift = stats + 768;

  // pack all weights (deterministic, every call)
  for (int blk = 0; blk < 3; ++blk)
    for (int l = 0; l < 4; ++l) {
      int tot = Ndim[l] * Kdim[blk][l];
      pack_weights<<<(tot + 255) / 256, 256, 0, stream>>>(W[blk][l], pw[blk][l], Ndim[l], Kdim[blk][l]);
    }

  auto zero = [&](float* p, long long n) {
    zero_f32<<<(unsigned)((n + 255) / 256), 256, 0, stream>>>(p, n);
  };

  // ---------------- edge block (B = E, K0 = 512) ----------------
  {
    int gb = (E + 63) / 64;
    float invB = 1.f / (float)E;
    zero(stSum, 512);
    gemm_mlp<512, 256, 0, true><<<gb, 256, 0, stream>>>(
        pw[BE][0], bb[BE][0], E, nullptr, nullptr, nullptr,
        globalF, nodeF, nullptr, edgeF, conn, gie, HA, nullptr, stSum, stSq);
    norm_params<<<1, 256, 0, stream>>>(stSum, stSq, gaP[BE][0], beP[BE][0], stScale, stShift, invB, 256);
    zero(stSum, 512);
    gemm_mlp<256, 256, 2, true><<<gb, 256, 0, stream>>>(
        pw[BE][1], bb[BE][1], E, HA, stScale, stShift,
        nullptr, nullptr, nullptr, nullptr, nullptr, nullptr, HB, nullptr, stSum, stSq);
    norm_params<<<1, 256, 0, stream>>>(stSum, stSq, gaP[BE][1], beP[BE][1], stScale, stShift, invB, 256);
    zero(stSum, 512);
    gemm_mlp<256, 256, 2, true><<<gb, 256, 0, stream>>>(
        pw[BE][2], bb[BE][2], E, HB, stScale, stShift,
        nullptr, nullptr, nullptr, nullptr, nullptr, nullptr, HA, nullptr, stSum, stSq);
    norm_params<<<1, 256, 0, stream>>>(stSum, stSq, gaP[BE][2], beP[BE][2], stScale, stShift, invB, 256);
    gemm_mlp<256, 128, 2, false><<<gb, 256, 0, stream>>>(
        pw[BE][3], bb[BE][3], E, HA, stScale, stShift,
        nullptr, nullptr, nullptr, nullptr, nullptr, nullptr, nullptr, outEdge, nullptr, nullptr);
  }

  // ---------------- aggregations from new_edge ----------------
  zero(aggNode, (long long)Nn * 128);
  {
    long long tot = (long long)E * 128;
    scatter_add_rows<<<(unsigned)((tot + 255) / 256), 256, 0, stream>>>(outEdge, conn, aggNode, tot);
    zero(aggEG, (long long)G * 128);
    scatter_add_graph<<<1024, 256, 0, stream>>>(outEdge, gie, aggEG, tot);
  }

  // ---------------- node block (B = Nn, K0 = 384) ----------------
  {
    int gb = (Nn + 63) / 64;
    float invB = 1.f / (float)Nn;
    zero(stSum, 512);
    gemm_mlp<384, 256, 1, true><<<gb, 256, 0, stream>>>(
        pw[BN][0], bb[BN][0], Nn, nullptr, nullptr, nullptr,
        globalF, nodeF, aggNode, nullptr, nullptr, gin, HA, nullptr, stSum, stSq);
    norm_params<<<1, 256, 0, stream>>>(stSum, stSq, gaP[BN][0], beP[BN][0], stScale, stShift, invB, 256);
    zero(stSum, 512);
    gemm_mlp<256, 256, 2, true><<<gb, 256, 0, stream>>>(
        pw[BN][1], bb[BN][1], Nn, HA, stScale, stShift,
        nullptr, nullptr, nullptr, nullptr, nullptr, nullptr, HB, nullptr, stSum, stSq);
    norm_params<<<1, 256, 0, stream>>>(stSum, stSq, gaP[BN][1], beP[BN][1], stScale, stShift, invB, 256);
    zero(stSum, 512);
    gemm_mlp<256, 256, 2, true><<<gb, 256, 0, stream>>>(
        pw[BN][2], bb[BN][2], Nn, HB, stScale, stShift,
        nullptr, nullptr, nullptr, nullptr, nullptr, nullptr, HA, nullptr, stSum, stSq);
    norm_params<<<1, 256, 0, stream>>>(stSum, stSq, gaP[BN][2], beP[BN][2], stScale, stShift, invB, 256);
    gemm_mlp<256, 128, 2, false><<<gb, 256, 0, stream>>>(
        pw[BN][3], bb[BN][3], Nn, HA, stScale, stShift,
        nullptr, nullptr, nullptr, nullptr, nullptr, nullptr, nullptr, outNode, nullptr, nullptr);
  }

  // ---------------- node -> graph aggregation ----------------
  zero(aggNG, (long long)G * 128);
  {
    long long tot = (long long)Nn * 128;
    scatter_add_graph<<<256, 256, 0, stream>>>(outNode, gin, aggNG, tot);
  }

  // ---------------- global block (B = G, K0 = 384) ----------------
  {
    int gb = (G + 63) / 64;
    float invB = 1.f / (float)G;
    zero(stSum, 512);
    gemm_mlp<384, 256, 1, true><<<gb, 256, 0, stream>>>(
        pw[BG][0], bb[BG][0], G, nullptr, nullptr, nullptr,
        globalF, aggNG, aggEG, nullptr, nullptr, nullptr, HA, nullptr, stSum, stSq);
    norm_params<<<1, 256, 0, stream>>>(stSum, stSq, gaP[BG][0], beP[BG][0], stScale, stShift, invB, 256);
    zero(stSum, 512);
    gemm_mlp<256, 256, 2, true><<<gb, 256, 0, stream>>>(
        pw[BG][1], bb[BG][1], G, HA, stScale, stShift,
        nullptr, nullptr, nullptr, nullptr, nullptr, nullptr, HB, nullptr, stSum, stSq);
    norm_params<<<1, 256, 0, stream>>>(stSum, stSq, gaP[BG][1], beP[BG][1], stScale, stShift, invB, 256);
    zero(stSum, 512);
    gemm_mlp<256, 256, 2, true><<<gb, 256, 0, stream>>>(
        pw[BG][2], bb[BG][2], G, HB, stScale, stShift,
        nullptr, nullptr, nullptr, nullptr, nullptr, nullptr, HA, nullptr, stSum, stSq);
    norm_params<<<1, 256, 0, stream>>>(stSum, stSq, gaP[BG][2], beP[BG][2], stScale, stShift, invB, 256);
    gemm_mlp<256, 128, 2, false><<<gb, 256, 0, stream>>>(
        pw[BG][3], bb[BG][3], G, HA, stScale, stShift,
        nullptr, nullptr, nullptr, nullptr, nullptr, nullptr, nullptr, outGlob, nullptr, nullptr);
  }
}
